// GcnPromptLearner_63307817943452
// MI455X (gfx1250) — matrix operand order repaired and checked by reading, not compiled, not run on hardware
//
#include <hip/hip_runtime.h>
#include <cstdint>
#include <cstddef>

// ---------------------------------------------------------------------------
// GCN 2-layer forward for MI455X (gfx1250, wave32, WMMA).
//   h1 = scatter_norm(x @ W1) + b1 ; relu
//   out = scatter_norm(h1 @ W2) + b2
// GEMMs run on v_wmma_f32_16x16x32_bf16 (bf16 in, f32 accum); each wave owns a
// 16x64 C tile so one A fragment feeds 4 WMMAs (2.5 b128 loads per WMMA).
// Aggregation runs on float4 gathers + global_atomic_add_f32 scatters.
// ---------------------------------------------------------------------------

#define N_NODES 50000
#define N_EDGES 150000
#define DIM     512
#define TOTAL_ELEMS ((size_t)N_NODES * DIM)   // 25,600,000

typedef __attribute__((ext_vector_type(8)))  __bf16 v8bf;
typedef __attribute__((ext_vector_type(16))) __bf16 v16bf;
typedef __attribute__((ext_vector_type(8)))  float  v8f;

__device__ __forceinline__ unsigned short f2bf(float f) {
  union { float f; unsigned int u; } v; v.f = f;
  unsigned int u = v.u;
  unsigned int r = u + 0x7FFFu + ((u >> 16) & 1u);   // round-to-nearest-even
  return (unsigned short)(r >> 16);
}

// ---- degree accumulation (self-loops folded into finalize) -----------------
__global__ __launch_bounds__(256)
void deg_kernel(const long long* __restrict__ ei, float* __restrict__ deg) {
  int e = blockIdx.x * blockDim.x + threadIdx.x;
  if (e < N_EDGES) atomicAdd(&deg[(int)ei[N_EDGES + e]], 1.0f);
}

__global__ __launch_bounds__(256)
void dinv_kernel(float* __restrict__ deg) {
  int i = blockIdx.x * blockDim.x + threadIdx.x;
  if (i < N_NODES) deg[i] = rsqrtf(deg[i] + 1.0f);   // deg >= 1 always
}

// ---- fp32 -> bf16 conversion ----------------------------------------------
__global__ __launch_bounds__(256)
void cvt_bf16_kernel(const float* __restrict__ in, unsigned short* __restrict__ out) {
  size_t i = (size_t)blockIdx.x * blockDim.x + threadIdx.x;
  if (i < TOTAL_ELEMS) out[i] = f2bf(in[i]);
}

// W [k, n] row-major fp32  ->  Wt [n, k] row-major bf16 (so B fragments are
// contiguous along K and load as b128).
__global__ __launch_bounds__(256)
void cvt_w_transpose_kernel(const float* __restrict__ W, unsigned short* __restrict__ Wt) {
  int idx = blockIdx.x * blockDim.x + threadIdx.x;      // over DIM*DIM
  if (idx < DIM * DIM) {
    int n = idx >> 9;          // output row   (original column)
    int k = idx & (DIM - 1);   // output col   (original row)
    Wt[idx] = f2bf(W[k * DIM + n]);
  }
}

// ---- bf16 WMMA GEMM: C[M,N] = A[M,K] * Bt[N,K]^T ---------------------------
// One wave computes one 16x64 C tile (4 x 16x16 accumulators), reusing one A
// fragment across 4 WMMAs per K-step. 4 waves per block along N.
// A-fragment lane layout (16x32 bf16): lane L holds row M=L&15;
//   khalf=(L>>4)*16; elements 0..15 = K khalf..khalf+15.
// B-fragment lane layout (32x16 bf16): lane L holds col N=L&15;
//   kbase=(L>>4)*16; elements 0..15 = K kbase..kbase+15 (contiguous in Wt).
__global__ __launch_bounds__(128)
void gemm_bf16_wmma_kernel(const unsigned short* __restrict__ A,   // [N_NODES, DIM] bf16
                           const unsigned short* __restrict__ Bt,  // [DIM, DIM] bf16 (W^T)
                           float* __restrict__ C) {                // [N_NODES, DIM] f32
  const int lane   = threadIdx.x & 31;
  const int wave   = threadIdx.x >> 5;
  const int tile64 = blockIdx.x * 4 + wave;            // 0..7  (64-wide N tile)
  const int tileM  = blockIdx.y;                       // 0..3124
  const int l15    = lane & 15;
  const int hi     = lane >> 4;                        // 0 or 1
  const int khalf  = hi << 4;                          // A: 0 or 16
  const int kb16   = hi << 4;                          // B: 0 or 16
  const int n0     = tile64 << 6;                      // first of 64 columns

  const unsigned short* arow = A + (size_t)((tileM << 4) + l15) * DIM;
  const unsigned short* b0p  = Bt + (size_t)(n0 + 0  + l15) * DIM + kb16;
  const unsigned short* b1p  = Bt + (size_t)(n0 + 16 + l15) * DIM + kb16;
  const unsigned short* b2p  = Bt + (size_t)(n0 + 32 + l15) * DIM + kb16;
  const unsigned short* b3p  = Bt + (size_t)(n0 + 48 + l15) * DIM + kb16;

  v8f c0 = {}, c1 = {}, c2 = {}, c3 = {};
#pragma unroll 2
  for (int k0 = 0; k0 < DIM; k0 += 32) {
    v8bf aL = *(const v8bf*)(arow + k0 + khalf);        // K: khalf .. +7
    v8bf aH = *(const v8bf*)(arow + k0 + khalf + 8);    // K: khalf+8 .. +15
    v16bf a = __builtin_shufflevector(aL, aH, 0,1,2,3,4,5,6,7,8,9,10,11,12,13,14,15);

    v8bf q0 = *(const v8bf*)(b0p + k0);
    v8bf q1 = *(const v8bf*)(b0p + k0 + 8);
    v16bf b0 = __builtin_shufflevector(q0, q1, 0,1,2,3,4,5,6,7,8,9,10,11,12,13,14,15);
    c0 = __builtin_amdgcn_wmma_f32_16x16x32_bf16(false, a, false, b0, (short)0, c0, false, false);

    v8bf q2 = *(const v8bf*)(b1p + k0);
    v8bf q3 = *(const v8bf*)(b1p + k0 + 8);
    v16bf b1 = __builtin_shufflevector(q2, q3, 0,1,2,3,4,5,6,7,8,9,10,11,12,13,14,15);
    c1 = __builtin_amdgcn_wmma_f32_16x16x32_bf16(false, a, false, b1, (short)0, c1, false, false);

    v8bf q4 = *(const v8bf*)(b2p + k0);
    v8bf q5 = *(const v8bf*)(b2p + k0 + 8);
    v16bf b2 = __builtin_shufflevector(q4, q5, 0,1,2,3,4,5,6,7,8,9,10,11,12,13,14,15);
    c2 = __builtin_amdgcn_wmma_f32_16x16x32_bf16(false, a, false, b2, (short)0, c2, false, false);

    v8bf q6 = *(const v8bf*)(b3p + k0);
    v8bf q7 = *(const v8bf*)(b3p + k0 + 8);
    v16bf b3 = __builtin_shufflevector(q6, q7, 0,1,2,3,4,5,6,7,8,9,10,11,12,13,14,15);
    c3 = __builtin_amdgcn_wmma_f32_16x16x32_bf16(false, a, false, b3, (short)0, c3, false, false);
  }

  // C layout: VGPR r -> row tileM*16 + hi*8 + r, col n0 + j*16 + l15.
  float* crow = C + (size_t)((tileM << 4) + (hi << 3)) * DIM + n0 + l15;
#pragma unroll
  for (int r = 0; r < 8; ++r) {
    crow[(size_t)r * DIM + 0]  = c0[r];
    crow[(size_t)r * DIM + 16] = c1[r];
    crow[(size_t)r * DIM + 32] = c2[r];
    crow[(size_t)r * DIM + 48] = c3[r];
  }
}

// ---- normalized gather/scatter aggregation ---------------------------------
// Block e handles one edge (e < E: real edge; else self-loop of node e-E).
// 128 threads x float4 covers the 512-wide feature row.
__global__ __launch_bounds__(128)
void scatter_kernel(const long long* __restrict__ ei, const float* __restrict__ dinv,
                    const float* __restrict__ h, float* __restrict__ out) {
  int e = blockIdx.x;
  int s, d;
  if (e < N_EDGES) { s = (int)ei[e]; d = (int)ei[N_EDGES + e]; }
  else             { s = d = e - N_EDGES; }
  float w = dinv[s] * dinv[d];
  const float4* hs = (const float4*)(h + (size_t)s * DIM);
  float* od = out + (size_t)d * DIM;
  int t = threadIdx.x;
  float4 v = hs[t];
  atomicAdd(&od[4 * t + 0], v.x * w);
  atomicAdd(&od[4 * t + 1], v.y * w);
  atomicAdd(&od[4 * t + 2], v.z * w);
  atomicAdd(&od[4 * t + 3], v.w * w);
}

// ---- bias + relu -> bf16 (feeds second GEMM directly) ----------------------
__global__ __launch_bounds__(256)
void biasrelu_bf16_kernel(const float* __restrict__ agg, const float* __restrict__ bias,
                          unsigned short* __restrict__ out) {
  size_t i = (size_t)blockIdx.x * blockDim.x + threadIdx.x;
  if (i < TOTAL_ELEMS) {
    float v = agg[i] + bias[i & (DIM - 1)];
    out[i] = f2bf(v > 0.0f ? v : 0.0f);
  }
}

// ---- final bias add (in place) --------------------------------------------
__global__ __launch_bounds__(256)
void biasadd_kernel(float* __restrict__ out, const float* __restrict__ bias) {
  size_t i = (size_t)blockIdx.x * blockDim.x + threadIdx.x;
  if (i < TOTAL_ELEMS) out[i] += bias[i & (DIM - 1)];
}

// ---------------------------------------------------------------------------
extern "C" void kernel_launch(void* const* d_in, const int* in_sizes, int n_in,
                              void* d_out, int out_size, void* d_ws, size_t ws_size,
                              hipStream_t stream) {
  const float*     x  = (const float*)d_in[0];
  const long long* ei = (const long long*)d_in[1];   // int64 edge_index [2, E]
  const float*     W1 = (const float*)d_in[2];
  const float*     b1 = (const float*)d_in[3];
  const float*     W2 = (const float*)d_in[4];
  const float*     b2 = (const float*)d_in[5];
  float* out = (float*)d_out;

  // workspace carve-out (256B aligned)
  char* ws = (char*)d_ws;
  size_t off = 0;
  auto carve = [&](size_t bytes) -> void* {
    void* p = ws + off;
    off += (bytes + 255) & ~(size_t)255;
    return p;
  };
  float*          dinv = (float*)carve((size_t)N_NODES * sizeof(float));          // 200 KB
  float*          h    = (float*)carve(TOTAL_ELEMS * sizeof(float));              // 102 MB
  unsigned short* Abf  = (unsigned short*)carve(TOTAL_ELEMS * sizeof(unsigned short)); // 51 MB
  unsigned short* Wt   = (unsigned short*)carve((size_t)DIM * DIM * sizeof(unsigned short)); // 0.5 MB

  const int ETOT = N_EDGES + N_NODES;
  const dim3 gemmGrid(DIM / 64 / 4, N_NODES / 16);   // (2, 3125), 4 waves/block, 16x64 per wave

  // ---- degree / normalization ----
  hipMemsetAsync(dinv, 0, (size_t)N_NODES * sizeof(float), stream);
  deg_kernel<<<(N_EDGES + 255) / 256, 256, 0, stream>>>(ei, dinv);
  dinv_kernel<<<(N_NODES + 255) / 256, 256, 0, stream>>>(dinv);

  // ---- layer 1 ----
  cvt_bf16_kernel<<<(unsigned)((TOTAL_ELEMS + 255) / 256), 256, 0, stream>>>(x, Abf);
  cvt_w_transpose_kernel<<<(DIM * DIM + 255) / 256, 256, 0, stream>>>(W1, Wt);
  gemm_bf16_wmma_kernel<<<gemmGrid, 128, 0, stream>>>(Abf, Wt, h);
  hipMemsetAsync(out, 0, TOTAL_ELEMS * sizeof(float), stream);
  scatter_kernel<<<ETOT, 128, 0, stream>>>(ei, dinv, h, out);
  biasrelu_bf16_kernel<<<(unsigned)((TOTAL_ELEMS + 255) / 256), 256, 0, stream>>>(out, b1, Abf);

  // ---- layer 2 ----
  cvt_w_transpose_kernel<<<(DIM * DIM + 255) / 256, 256, 0, stream>>>(W2, Wt);
  gemm_bf16_wmma_kernel<<<gemmGrid, 128, 0, stream>>>(Abf, Wt, h);
  hipMemsetAsync(out, 0, TOTAL_ELEMS * sizeof(float), stream);
  scatter_kernel<<<ETOT, 128, 0, stream>>>(ei, dinv, h, out);
  biasadd_kernel<<<(unsigned)((TOTAL_ELEMS + 255) / 256), 256, 0, stream>>>(out, b2);
}
